// QuantumSumBlock_26688926777599
// MI455X (gfx1250) — compile-verified, matrix operand order checked by reading, not run on hardware
//
#include <hip/hip_runtime.h>

typedef __attribute__((ext_vector_type(16))) _Float16 v16h;
typedef __attribute__((ext_vector_type(8)))  float    v8f;

#define TPB      256   // 8 wave32 waves per block
#define NBLOCKS  1024  // fixed partial count -> deterministic reduction
#define TBLOCKS  64    // tail-kernel blocks (only launched when n % step != 0)

// v_cos_f32 takes its input in revolutions (1.0 == 2*pi radians) and performs
// hardware range reduction on the fractional part.
__device__ __forceinline__ float fast_cos(float x) {
  return __builtin_amdgcn_cosf(x * 0.15915494309189535f);
}

// Stage 1 (bulk): mask-free streaming pass over `iters` full grid sweeps.
// Per iteration each lane does 8 unconditional global_load_b128, 16 v_cos_f32,
// 16 RTE f16 converts, and one v_wmma_f32_16x16x32_f16 accumulate (all-ones B
// -> the matrix pipe performs 512 adds/wave/instr, co-executing with VALU).
__global__ __launch_bounds__(TPB) void qsum_main_kernel(
    const float4* __restrict__ phi4, const float4* __restrict__ w4,
    float* __restrict__ partial, unsigned int iters) {
  const unsigned int tid  = threadIdx.x;
  const unsigned int gtid = blockIdx.x * TPB + tid;
  const unsigned int NTH  = NBLOCKS * TPB;

  v16h ones;
#pragma unroll
  for (int i = 0; i < 16; ++i) ones[i] = (_Float16)1.0f;

  v8f acc = {0.f, 0.f, 0.f, 0.f, 0.f, 0.f, 0.f, 0.f};

  const float4* __restrict__ p = phi4 + gtid;
  const float4* __restrict__ q = w4 + gtid;

  for (unsigned int it = 0; it < iters; ++it) {
    union { v16h v; _Float16 h[16]; } a;
#pragma unroll
    for (int j = 0; j < 4; ++j) {
      float4 pv = p[(size_t)j * NTH];   // fully coalesced b128 loads
      float4 wv = q[(size_t)j * NTH];
      a.h[4 * j + 0] = (_Float16)fast_cos(pv.x * wv.x);  // RTE cvt: unbiased
      a.h[4 * j + 1] = (_Float16)fast_cos(pv.y * wv.y);
      a.h[4 * j + 2] = (_Float16)fast_cos(pv.z * wv.z);
      a.h[4 * j + 3] = (_Float16)fast_cos(pv.w * wv.w);
    }
    // C += A x ones : every D column holds the A row-sums (16x replication)
    acc = __builtin_amdgcn_wmma_f32_16x16x32_f16(
        /*neg_a=*/false, a.v, /*neg_b=*/false, ones,
        /*c_mod=*/(short)0, acc, /*reuse_a=*/false, /*reuse_b=*/false);
    p += (size_t)4 * NTH;
    q += (size_t)4 * NTH;
  }

  // Per-lane sum of the 8 accumulator VGPRs, then deterministic LDS tree.
  float s = acc[0] + acc[1] + acc[2] + acc[3] + acc[4] + acc[5] + acc[6] + acc[7];
  __shared__ float sdata[TPB];
  sdata[tid] = s;
  __syncthreads();
#pragma unroll
  for (unsigned int off = TPB / 2; off > 0; off >>= 1) {
    if (tid < off) sdata[tid] += sdata[tid + off];
    __syncthreads();
  }
  // /16 here so bulk and tail partials are directly summable downstream.
  if (tid == 0) partial[blockIdx.x] = sdata[0] * 0.0625f;
}

// Stage 1b (tail): plain scalar grid-stride sum over [start, n).
// Only launched when n is not a multiple of the bulk sweep size.
__global__ __launch_bounds__(TPB) void qsum_tail_kernel(
    const float* __restrict__ phi, const float* __restrict__ w,
    float* __restrict__ partial, unsigned int start, unsigned int n) {
  float s = 0.0f;
  for (unsigned int i = start + blockIdx.x * TPB + threadIdx.x; i < n;
       i += TBLOCKS * TPB)
    s += fast_cos(phi[i] * w[i]);
  __shared__ float sdata[TPB];
  sdata[threadIdx.x] = s;
  __syncthreads();
#pragma unroll
  for (unsigned int off = TPB / 2; off > 0; off >>= 1) {
    if (threadIdx.x < off) sdata[threadIdx.x] += sdata[threadIdx.x + off];
    __syncthreads();
  }
  if (threadIdx.x == 0) partial[NBLOCKS + blockIdx.x] = sdata[0];
}

// Stage 2: single block reduces the partials in a fixed order and applies the
// closed-form (s + (M-n))/M.
__global__ __launch_bounds__(TPB) void qsum_final_kernel(
    const float* __restrict__ partial, float* __restrict__ out,
    unsigned int npart, float m_minus_n, float inv_m) {
  __shared__ float sdata[TPB];
  float s = 0.0f;
  for (unsigned int i = threadIdx.x; i < npart; i += TPB) s += partial[i];
  sdata[threadIdx.x] = s;
  __syncthreads();
#pragma unroll
  for (unsigned int off = TPB / 2; off > 0; off >>= 1) {
    if (threadIdx.x < off) sdata[threadIdx.x] += sdata[threadIdx.x + off];
    __syncthreads();
  }
  if (threadIdx.x == 0) out[0] = (sdata[0] + m_minus_n) * inv_m;
}

extern "C" void kernel_launch(void* const* d_in, const int* in_sizes, int n_in,
                              void* d_out, int out_size, void* d_ws, size_t ws_size,
                              hipStream_t stream) {
  const float* phi = (const float*)d_in[0];
  const float* w   = (const float*)d_in[1];
  float*       out = (float*)d_out;
  unsigned int n   = (unsigned int)in_sizes[0];

  // M = 2^ceil(log2(n)); for n = 2^24 this is exactly n, so M - n == 0.
  unsigned long long M = 1;
  while (M < (unsigned long long)n) M <<= 1;
  float m_minus_n = (float)(double)(M - (unsigned long long)n);
  float inv_m     = (float)(1.0 / (double)M);

  const unsigned int step  = NBLOCKS * TPB * 16u;  // elements per full sweep
  const unsigned int iters = n / step;             // 4 for n = 2^24
  const unsigned int start = iters * step;

  float* partial = (float*)d_ws;  // up to (NBLOCKS + TBLOCKS) * 4 bytes

  qsum_main_kernel<<<NBLOCKS, TPB, 0, stream>>>(
      (const float4*)phi, (const float4*)w, partial, iters);

  unsigned int npart = NBLOCKS;
  if (start < n) {  // depends only on input size -> deterministic
    qsum_tail_kernel<<<TBLOCKS, TPB, 0, stream>>>(phi, w, partial, start, n);
    npart += TBLOCKS;
  }

  qsum_final_kernel<<<1, TPB, 0, stream>>>(partial, out, npart, m_minus_n, inv_m);
}
